// Block_89240830476883
// MI455X (gfx1250) — 8-bit matrix operand order repaired and checked by reading, compiled, not run on hardware
//
#include <hip/hip_runtime.h>
#include <hip/hip_bf16.h>

typedef int v8i __attribute__((ext_vector_type(8)));

#define NPIX   100352
#define CIN    64
#define P      384
#define COUT   96
#define HW     3136
#define BN_EPS 1e-5f

__device__ __forceinline__ float pow2scale(float m, float n) {
    m = fmaxf(m, 1e-8f);
    return exp2f(ceilf(log2f(m / n)));
}

__global__ void k_init(unsigned* U) {
    if (threadIdx.x < 16) U[threadIdx.x] = 0u;
}

__global__ void k_absmax(const float* __restrict__ src, int n, unsigned* U, int uidx) {
    __shared__ unsigned sm;
    if (threadIdx.x == 0) sm = 0u;
    __syncthreads();
    float local = 0.0f;
    for (int i = blockIdx.x * blockDim.x + threadIdx.x; i < n; i += gridDim.x * blockDim.x)
        local = fmaxf(local, fabsf(src[i]));
    atomicMax(&sm, __float_as_uint(local));
    __syncthreads();
    if (threadIdx.x == 0) atomicMax(&U[uidx], sm);
}

__global__ void k_scales0(float* S, const unsigned* U) {
    if (threadIdx.x == 0 && blockIdx.x == 0) {
        S[0] = pow2scale(__uint_as_float(U[0]), 7.0f);
        S[1] = pow2scale(__uint_as_float(U[1]), 7.0f);
        S[2] = pow2scale(__uint_as_float(U[2]), 7.0f);
        S[3] = pow2scale(__uint_as_float(U[3]), 7.0f);
        S[4] = pow2scale(__uint_as_float(U[4]), 7.0f);
    }
}
__global__ void k_scale_code(float* S, const unsigned* U, int uidx, int sidx, float n) {
    if (threadIdx.x == 0 && blockIdx.x == 0)
        S[sidx] = pow2scale(0.25f * (float)U[uidx], n);
}
__global__ void k_scale_out(float* S, const unsigned* U, int uidx, int sa, int sb, int sidx, float n) {
    if (threadIdx.x == 0 && blockIdx.x == 0)
        S[sidx] = pow2scale((float)U[uidx] * S[sa] * S[sb], n);
}

__global__ void k_bnprep(const float* __restrict__ g1, const float* __restrict__ b1,
                         const float* __restrict__ m1, const float* __restrict__ v1,
                         const float* __restrict__ g2, const float* __restrict__ b2,
                         const float* __restrict__ m2, const float* __restrict__ v2,
                         const float* __restrict__ g3, const float* __restrict__ b3,
                         const float* __restrict__ m3, const float* __restrict__ v3,
                         const float* __restrict__ gs, const float* __restrict__ bs,
                         const float* __restrict__ ms, const float* __restrict__ vs,
                         float* __restrict__ BN) {
    int i = blockIdx.x * blockDim.x + threadIdx.x;
    if (i < P) {
        float inv = g1[i] * rsqrtf(v1[i] + BN_EPS);
        BN[i] = inv;  BN[P + i] = b1[i] - m1[i] * inv;
        float inv2 = g2[i] * rsqrtf(v2[i] + BN_EPS);
        BN[2 * P + i] = inv2;  BN[3 * P + i] = b2[i] - m2[i] * inv2;
    }
    if (i < COUT) {
        float inv3 = g3[i] * rsqrtf(v3[i] + BN_EPS);
        BN[4 * P + i] = inv3;  BN[4 * P + COUT + i] = b3[i] - m3[i] * inv3;
        float invs = gs[i] * rsqrtf(vs[i] + BN_EPS);
        BN[4 * P + 2 * COUT + i] = invs;  BN[4 * P + 3 * COUT + i] = bs[i] - ms[i] * invs;
    }
}

__global__ void k_quant(const float* __restrict__ src, int8_t* __restrict__ dst, int n,
                        const float* S, int sidx) {
    int i = blockIdx.x * blockDim.x + threadIdx.x;
    if (i < n) {
        float r = 1.0f / S[sidx];
        float q = fminf(fmaxf(rintf(src[i] * r), -8.0f), 7.0f);
        dst[i] = (int8_t)(int)q;
    }
}

__global__ __launch_bounds__(256) void k_qx(const float* __restrict__ x,
                                            int8_t* __restrict__ Xq, const float* S) {
    __shared__ float tile[64][65];
    int p0  = blockIdx.x << 6;
    int img = p0 / HW, hw0 = p0 - img * HW;
    int t = threadIdx.x;
#pragma unroll
    for (int it = 0; it < 16; it++) {
        int c   = (it << 2) + (t >> 6);
        int hwl = t & 63;
        tile[c][hwl] = x[(size_t)(img * CIN + c) * HW + hw0 + hwl];
    }
    __syncthreads();
    float r = 1.0f / S[0];
    int* Xo = (int*)Xq;
#pragma unroll
    for (int it = 0; it < 4; it++) {
        int idx = (it << 8) + t;
        int hwl = idx >> 4;
        int c4  = idx & 15;
        unsigned pk = 0;
#pragma unroll
        for (int j = 0; j < 4; j++) {
            float q = fminf(fmaxf(rintf(tile[c4 * 4 + j][hwl] * r), -8.0f), 7.0f);
            pk |= ((unsigned)((int)q) & 0xFF) << (8 * j);
        }
        Xo[(p0 + hwl) * (CIN / 4) + c4] = (int)pk;
    }
}

__global__ __launch_bounds__(256) void k_conv1(const int8_t* __restrict__ Xq,
                                               const int8_t* __restrict__ Wq,
                                               const float* __restrict__ S,
                                               const float* __restrict__ bni,
                                               const float* __restrict__ bnb,
                                               uint8_t* __restrict__ Y1, unsigned* U) {
    __shared__ unsigned sm;
    if (threadIdx.x == 0) sm = 0u;
    __syncthreads();

    int wave = (blockIdx.x * blockDim.x + threadIdx.x) >> 5;
    int lane = threadIdx.x & 31;
    int half = lane >> 4, ln = lane & 15;
    int row0 = wave << 4;

    const int* Xi = (const int*)Xq;
    const int* Wi = (const int*)Wq;

    v8i A;
    int abase = (row0 + ln) << 4;
#pragma unroll
    for (int t = 0; t < 8; t++)
        A[t] = Xi[abase + ((t >> 1) << 2) + (half << 1) + (t & 1)];

    float sxw = S[0] * S[1];
    unsigned lm = 0;

    auto loadB = [&](int nt) {
        v8i B;
        int col = (nt << 4) + ln;
#pragma unroll
        for (int t = 0; t < 8; t++)
            B[t] = Wi[(col << 4) + ((t >> 1) << 2) + (half << 1) + (t & 1)];
        return B;
    };
    auto epi = [&](v8i C, int nt) {
        int col  = (nt << 4) + ln;
        float a    = sxw * bni[col];
        float beta = bnb[col];
#pragma unroll
        for (int r = 0; r < 8; r++) {
            float f = (float)C[r] * a + beta;
            unsigned code = (unsigned)fminf(fmaxf(rintf(f * 4.0f), 0.0f), 15.0f);
            lm = lm > code ? lm : code;
            Y1[(row0 + r + 8 * half) * P + col] = (uint8_t)code;
        }
    };

    v8i B = loadB(0);
    v8i C = __builtin_amdgcn_wmma_i32_16x16x64_iu8(true, A, true, B,
                                                   (v8i){0,0,0,0,0,0,0,0}, false, false);
#pragma unroll 1
    for (int nt = 1; nt < 24; nt++) {
        v8i B2 = loadB(nt);
        v8i C2 = __builtin_amdgcn_wmma_i32_16x16x64_iu8(true, A, true, B2,
                                                        (v8i){0,0,0,0,0,0,0,0}, false, false);
        epi(C, nt - 1);
        C = C2;
    }
    epi(C, 23);

    atomicMax(&sm, lm);
    __syncthreads();
    if (threadIdx.x == 0) atomicMax(&U[5], sm);
}

__global__ void k_dw(const uint8_t* __restrict__ Y1, const int8_t* __restrict__ W2q,
                     const float* __restrict__ S,
                     const float* __restrict__ bni,
                     const float* __restrict__ bnb,
                     uint8_t* __restrict__ Y2, unsigned* U) {
    __shared__ unsigned sm;
    if (threadIdx.x == 0) sm = 0u;
    __syncthreads();

    int i = blockIdx.x * blockDim.x + threadIdx.x;
    unsigned lm = 0;
    if (i < NPIX * (P / 4)) {
        int c4 = i % (P / 4);
        int p  = i / (P / 4);
        int img = p / HW, hw = p - img * HW;
        int h = hw / 56, w = hw - h * 56;
        float r2 = 0.25f / S[5];

        int wreg[4][9];
#pragma unroll
        for (int j = 0; j < 4; j++)
#pragma unroll
            for (int tp = 0; tp < 9; tp++)
                wreg[j][tp] = (int)W2q[(c4 * 4 + j) * 9 + tp];

        int acc[4] = {0, 0, 0, 0};
        const uint8_t* base = Y1 + (size_t)img * HW * P + c4 * 4;
#pragma unroll
        for (int dy = -1; dy <= 1; dy++) {
#pragma unroll
            for (int dx = -1; dx <= 1; dx++) {
                int hh = h + dy, ww = w + dx;
                if (hh >= 0 && hh < 56 && ww >= 0 && ww < 56) {
                    unsigned d = *(const unsigned*)(base + (size_t)(hh * 56 + ww) * P);
                    int tp = (dy + 1) * 3 + (dx + 1);
#pragma unroll
                    for (int j = 0; j < 4; j++) {
                        int code = (d >> (8 * j)) & 0xFF;
                        int xi = (int)fminf(fmaxf(rintf((float)code * r2), -8.0f), 7.0f);
                        acc[j] += xi * wreg[j][tp];
                    }
                }
            }
        }
        float4 iv = ((const float4*)bni)[c4];
        float4 bt = ((const float4*)bnb)[c4];
        float s22 = S[5] * S[2];
        float av[4] = {s22 * iv.x, s22 * iv.y, s22 * iv.z, s22 * iv.w};
        float bv[4] = {bt.x, bt.y, bt.z, bt.w};
        unsigned pk = 0;
#pragma unroll
        for (int j = 0; j < 4; j++) {
            float f = (float)acc[j] * av[j] + bv[j];
            unsigned code = (unsigned)fminf(fmaxf(rintf(f * 4.0f), 0.0f), 15.0f);
            lm = lm > code ? lm : code;
            pk |= code << (8 * j);
        }
        *(unsigned*)(Y2 + (size_t)p * P + c4 * 4) = pk;
    }
    atomicMax(&sm, lm);
    __syncthreads();
    if (threadIdx.x == 0) atomicMax(&U[6], sm);
}

__global__ void k_requant3(const uint8_t* __restrict__ Y2, int8_t* __restrict__ X3q,
                           const float* S) {
    int i = blockIdx.x * blockDim.x + threadIdx.x;
    if (i < NPIX * (P / 4)) {
        float r3 = 0.25f / S[6];
        unsigned d = ((const unsigned*)Y2)[i];
        unsigned pk = 0;
#pragma unroll
        for (int j = 0; j < 4; j++) {
            int code = (d >> (8 * j)) & 0xFF;
            float q = fminf(fmaxf(rintf((float)code * r3), -8.0f), 7.0f);
            pk |= ((unsigned)((int)q) & 0xFF) << (8 * j);
        }
        ((unsigned*)X3q)[i] = pk;
    }
}

template <int KS, int NT>
__global__ __launch_bounds__(256) void k_gemm_acc(const int8_t* __restrict__ Xq,
                                                  const int8_t* __restrict__ Wq,
                                                  int* __restrict__ Acc, unsigned* U, int uidx) {
    __shared__ unsigned sm;
    if (threadIdx.x == 0) sm = 0u;
    __syncthreads();

    const int rowdw = KS * 16;
    int wave = (blockIdx.x * blockDim.x + threadIdx.x) >> 5;
    int lane = threadIdx.x & 31;
    int half = lane >> 4, ln = lane & 15;
    int row0 = wave << 4;

    const int* Xi = (const int*)Xq;
    const int* Wi = (const int*)Wq;

    v8i A[KS];
    int abase = (row0 + ln) * rowdw;
#pragma unroll
    for (int kt = 0; kt < KS; kt++)
#pragma unroll
        for (int t = 0; t < 8; t++)
            A[kt][t] = Xi[abase + kt * 16 + ((t >> 1) << 2) + (half << 1) + (t & 1)];

    unsigned lm = 0;

    auto tileC = [&](int nt) {
        int col = (nt << 4) + ln;
        v8i C = {0, 0, 0, 0, 0, 0, 0, 0};
#pragma unroll
        for (int kt = 0; kt < KS; kt++) {
            v8i B;
#pragma unroll
            for (int t = 0; t < 8; t++)
                B[t] = Wi[col * rowdw + kt * 16 + ((t >> 1) << 2) + (half << 1) + (t & 1)];
            C = __builtin_amdgcn_wmma_i32_16x16x64_iu8(true, A[kt], true, B, C, false, false);
        }
        return C;
    };
    auto epi = [&](v8i C, int nt) {
        int col = (nt << 4) + ln;
#pragma unroll
        for (int r = 0; r < 8; r++) {
            int val = C[r];
            unsigned a = (unsigned)(val < 0 ? -val : val);
            lm = lm > a ? lm : a;
            Acc[(row0 + r + 8 * half) * (NT * 16) + col] = val;
        }
    };

    v8i C = tileC(0);
#pragma unroll 1
    for (int nt = 1; nt < NT; nt++) {
        v8i C2 = tileC(nt);
        epi(C, nt - 1);
        C = C2;
    }
    epi(C, NT - 1);

    atomicMax(&sm, lm);
    __syncthreads();
    if (threadIdx.x == 0) atomicMax(&U[uidx], sm);
}

__global__ __launch_bounds__(256) void k_epi(const int* __restrict__ Acc3, const int* __restrict__ AccS,
                      const float* __restrict__ S,
                      const float* __restrict__ i3, const float* __restrict__ be3,
                      const float* __restrict__ iS, const float* __restrict__ beS,
                      float* __restrict__ out) {
    __shared__ float tile[COUT][65];
    int p0  = blockIdx.x << 6;
    int img = p0 / HW, hw0 = p0 - img * HW;
    int t = threadIdx.x;

    float s3w = S[6] * S[3];
    float ssw = S[0] * S[4];
    float so3 = S[7], sos = S[8];
    float iso3 = 1.0f / so3, isos = 1.0f / sos;

#pragma unroll 1
    for (int it = 0; it < 24; it++) {
        int idx = (it << 8) + t;
        int c  = idx % COUT;
        int pl = idx / COUT;
        int ai = (p0 + pl) * COUT + c;

        float f3 = fminf(fmaxf(rintf((float)Acc3[ai] * s3w * iso3), -128.0f), 127.0f) * so3;
        float y3 = f3 * i3[c] + be3[c];

        float fs = fminf(fmaxf(rintf((float)AccS[ai] * ssw * isos), -8.0f), 7.0f) * sos;
        float ysc = fs * iS[c] + beS[c];

        tile[c][pl] = fminf(fmaxf(rintf((y3 + ysc) * 4.0f), 0.0f), 15.0f) * 0.25f;
    }
    __syncthreads();
#pragma unroll 1
    for (int it = 0; it < 24; it++) {
        int idx = (it << 8) + t;
        int pl = idx & 63;
        int c  = idx >> 6;
        out[(size_t)(img * COUT + c) * HW + hw0 + pl] = tile[c][pl];
    }
}

extern "C" void kernel_launch(void* const* d_in, const int* in_sizes, int n_in,
                              void* d_out, int out_size, void* d_ws, size_t ws_size,
                              hipStream_t stream) {
    const float* x  = (const float*)d_in[0];
    const float* w1 = (const float*)d_in[1];
    const float *g1 = (const float*)d_in[2], *b1 = (const float*)d_in[3],
                *m1 = (const float*)d_in[4], *v1 = (const float*)d_in[5];
    const float* w2 = (const float*)d_in[6];
    const float *g2 = (const float*)d_in[7], *b2 = (const float*)d_in[8],
                *m2 = (const float*)d_in[9], *v2 = (const float*)d_in[10];
    const float* w3 = (const float*)d_in[11];
    const float *g3 = (const float*)d_in[12], *b3 = (const float*)d_in[13],
                *m3 = (const float*)d_in[14], *v3 = (const float*)d_in[15];
    const float* ws = (const float*)d_in[16];
    const float *gs = (const float*)d_in[17], *bs = (const float*)d_in[18],
                *ms = (const float*)d_in[19], *vs = (const float*)d_in[20];
    float* out = (float*)d_out;

    char* w = (char*)d_ws;
    float*    S   = (float*)w;
    unsigned* U   = (unsigned*)(w + 64);
    int8_t*   W1q = (int8_t*)(w + 256);
    int8_t*   W2q = W1q + P * CIN;
    int8_t*   W3q = W2q + P * 9;
    int8_t*   WSq = W3q + COUT * P;
    float*    BN  = (float*)(w + 80 * 1024);
    int8_t*   Xq  = (int8_t*)(w + 128 * 1024);
    uint8_t*  Y1  = (uint8_t*)(Xq + (size_t)NPIX * CIN);
    uint8_t*  Y2  = Y1 + (size_t)NPIX * P;
    int*      Acc3 = (int*)(Y2 + (size_t)NPIX * P);
    int*      AccS = (int*)((char*)Acc3 + (size_t)NPIX * COUT * 4);
    int8_t*   X3q = (int8_t*)Y1;

    float* inv1 = BN,            *beta1 = BN + P;
    float* inv2 = BN + 2 * P,    *beta2 = BN + 3 * P;
    float* inv3 = BN + 4 * P,    *beta3 = BN + 4 * P + COUT;
    float* invS = BN + 4 * P + 2 * COUT, *betaS = BN + 4 * P + 3 * COUT;

    const int T = 256;

    k_init<<<1, 32, 0, stream>>>(U);
    k_absmax<<<1024, T, 0, stream>>>(x,  NPIX * CIN, U, 0);
    k_absmax<<<96,   T, 0, stream>>>(w1, P * CIN,    U, 1);
    k_absmax<<<14,   T, 0, stream>>>(w2, P * 9,      U, 2);
    k_absmax<<<144,  T, 0, stream>>>(w3, COUT * P,   U, 3);
    k_absmax<<<24,   T, 0, stream>>>(ws, COUT * CIN, U, 4);
    k_scales0<<<1, 1, 0, stream>>>(S, U);
    k_bnprep<<<2, T, 0, stream>>>(g1, b1, m1, v1, g2, b2, m2, v2,
                                  g3, b3, m3, v3, gs, bs, ms, vs, BN);

    k_qx<<<NPIX / 64, T, 0, stream>>>(x, Xq, S);
    k_quant<<<(P * CIN + T - 1) / T,    T, 0, stream>>>(w1, W1q, P * CIN,    S, 1);
    k_quant<<<(P * 9 + T - 1) / T,      T, 0, stream>>>(w2, W2q, P * 9,      S, 2);
    k_quant<<<(COUT * P + T - 1) / T,   T, 0, stream>>>(w3, W3q, COUT * P,   S, 3);
    k_quant<<<(COUT * CIN + T - 1) / T, T, 0, stream>>>(ws, WSq, COUT * CIN, S, 4);

    k_conv1<<<784, T, 0, stream>>>(Xq, W1q, S, inv1, beta1, Y1, U);
    k_scale_code<<<1, 1, 0, stream>>>(S, U, 5, 5, 7.0f);

    k_dw<<<(NPIX * (P / 4) + T - 1) / T, T, 0, stream>>>(Y1, W2q, S, inv2, beta2, Y2, U);
    k_scale_code<<<1, 1, 0, stream>>>(S, U, 6, 6, 7.0f);

    k_requant3<<<(NPIX * (P / 4) + T - 1) / T, T, 0, stream>>>(Y2, X3q, S);
    k_gemm_acc<6, 6><<<784, T, 0, stream>>>(X3q, W3q, Acc3, U, 7);
    k_gemm_acc<1, 6><<<784, T, 0, stream>>>(Xq,  WSq, AccS, U, 8);

    k_scale_out<<<1, 1, 0, stream>>>(S, U, 7, 6, 3, 7, 127.0f);
    k_scale_out<<<1, 1, 0, stream>>>(S, U, 8, 0, 4, 8, 7.0f);

    k_epi<<<NPIX / 64, T, 0, stream>>>(Acc3, AccS, S,
                                       inv3, beta3, invS, betaS, out);
}
